// Dots1MoE_48636209660326
// MI455X (gfx1250) — matrix operand order repaired and checked by reading, not compiled, not run on hardware
//
#include <hip/hip_runtime.h>

// ---------------------------------------------------------------------------
// Dots1 MoE forward for MI455X (gfx1250, wave32, WMMA).
//   - bf16 WMMA (v_wmma_f32_16x16x32_bf16), f32 accumulate
//   - GLOBAL_LOAD_ASYNC_TO_LDS_B128 + double-buffered LDS pipeline
//   - fused gate_up GEMM + SiLU*mul*route_w epilogue (no f32 intermediate)
//   - single down-proj kernel: registers accumulate shared + 16 experts,
//     one deterministic write of out (no atomics, no RMW chains)
// ---------------------------------------------------------------------------

typedef __attribute__((ext_vector_type(16))) __bf16 v16bf;
typedef __attribute__((ext_vector_type(8)))  __bf16 v8bf;
typedef __attribute__((ext_vector_type(8)))  float  v8f;
typedef __attribute__((ext_vector_type(4)))  float  v4f;
typedef int v4i __attribute__((vector_size(16)));   // matches builtin param type

#define BK 32
#define LDA_PAD 40   // bf16 elems per LDS row (80 B: 16B-aligned, bank-skewed)
#define LDB_PAD 40

// ---- CDNA5 async global->LDS copy (ASYNCcnt), with safe fallback ----------
#if defined(__gfx1250__) && __has_builtin(__builtin_amdgcn_global_load_async_to_lds_b128)
#define HAVE_ASYNC_LDS 1
#endif

__device__ __forceinline__ void cp_async_b128(const __bf16* g, __bf16* l) {
#if HAVE_ASYNC_LDS
    __builtin_amdgcn_global_load_async_to_lds_b128(
        (__attribute__((address_space(1))) v4i*)g,
        (__attribute__((address_space(3))) v4i*)l, 0, 0);
#else
    *(v8bf*)l = *(const v8bf*)g;
#endif
}

__device__ __forceinline__ void wait_async_lds() {
#if HAVE_ASYNC_LDS
#if __has_builtin(__builtin_amdgcn_s_wait_asynccnt)
    __builtin_amdgcn_s_wait_asynccnt(0);
#else
    asm volatile("s_wait_asynccnt 0x0" ::: "memory");
#endif
#endif
}

// f32x16 (4x v4f) -> two v8bf
__device__ __forceinline__ void cvt16_bf16(const v4f b[4], v8bf& lo, v8bf& hi) {
    lo[0] = (__bf16)b[0][0]; lo[1] = (__bf16)b[0][1];
    lo[2] = (__bf16)b[0][2]; lo[3] = (__bf16)b[0][3];
    lo[4] = (__bf16)b[1][0]; lo[5] = (__bf16)b[1][1];
    lo[6] = (__bf16)b[1][2]; lo[7] = (__bf16)b[1][3];
    hi[0] = (__bf16)b[2][0]; hi[1] = (__bf16)b[2][1];
    hi[2] = (__bf16)b[2][2]; hi[3] = (__bf16)b[2][3];
    hi[4] = (__bf16)b[3][0]; hi[5] = (__bf16)b[3][1];
    hi[6] = (__bf16)b[3][2]; hi[7] = (__bf16)b[3][3];
}

// -------------------------------- convert ----------------------------------
__global__ __launch_bounds__(256)
void k_convert(const float* __restrict__ x, __bf16* __restrict__ y, int n) {
    int i = blockIdx.x * 256 + threadIdx.x;
    if (i < n) y[i] = (__bf16)x[i];
}

// -------------------------------- routing ----------------------------------
__global__ __launch_bounds__(32)
void k_route(const float* __restrict__ X, const float* __restrict__ Gw,
             const float* __restrict__ bias, float* __restrict__ route_w,
             int T, int H) {
    const int t = blockIdx.x;
    const int lane = threadIdx.x;
    float acc[16];
#pragma unroll
    for (int e = 0; e < 16; ++e) acc[e] = 0.f;
    const float* xrow = X + (size_t)t * H;
    for (int h = lane; h < H; h += 32) {
        float xv = xrow[h];
#pragma unroll
        for (int e = 0; e < 16; ++e) acc[e] += xv * Gw[(size_t)e * H + h];
    }
#pragma unroll
    for (int e = 0; e < 16; ++e) {
#pragma unroll
        for (int off = 16; off > 0; off >>= 1)
            acc[e] += __shfl_xor(acc[e], off, 32);
    }
    if (lane == 0) {
        float score[16], choice[16];
#pragma unroll
        for (int e = 0; e < 16; ++e) {
            score[e]  = 1.f / (1.f + __expf(-acc[e]));
            choice[e] = score[e] + bias[e];
        }
        bool used[16] = {};
        int  sel[6];
        float wsum = 0.f;
#pragma unroll
        for (int k = 0; k < 6; ++k) {
            int best = 0; float bv = -3.0e38f;
#pragma unroll
            for (int e = 0; e < 16; ++e)
                if (!used[e] && choice[e] > bv) { bv = choice[e]; best = e; }
            used[best] = true; sel[k] = best; wsum += score[best];
        }
        const float s = 2.5f / wsum;
        float* rw = route_w + (size_t)t * 16;
#pragma unroll
        for (int e = 0; e < 16; ++e) rw[e] = 0.f;
#pragma unroll
        for (int k = 0; k < 6; ++k) rw[sel[k]] = score[sel[k]] * s;
    }
}

// ------------------------- fused gate_up + SiLU*mul -------------------------
// grid.z: 0..15 = routed experts (Nout=I, scale=route_w), 16 = shared (Nout=Is).
// Block computes 128 rows x 64 cols of BOTH gate and up halves, then
// Out[t,n] = scale_t * silu(g) * u in bf16.  Double-buffered, async A tiles.
__global__ __launch_bounds__(256)
void k_gateup(const __bf16* __restrict__ Xb, const float* __restrict__ W1,
              const float* __restrict__ SW1, const float* __restrict__ route_w,
              __bf16* __restrict__ HbAll, __bf16* __restrict__ Hs,
              int T, int H, int I, int Is) {
    const int z = blockIdx.z;
    const bool se = (z == 16);
    const int Nout = se ? Is : I;
    const int n0 = blockIdx.x * 64;
    if (n0 >= Nout) return;
    const int m0 = blockIdx.y * 128;
    const int K = H;
    const float* Bw = se ? SW1 : (W1 + (size_t)z * (size_t)(2 * I) * H);
    __bf16* Out = se ? Hs : (HbAll + (size_t)z * (size_t)T * I);

    __shared__ __align__(16) __bf16 sA[2][128 * LDA_PAD];
    __shared__ __align__(16) __bf16 sB[2][2][64 * LDB_PAD];

    const int tid  = threadIdx.x;
    const int lane = tid & 31, wave = tid >> 5;
    const int wm = wave >> 1, wn = wave & 1;    // 4x2 wave grid
    const int ml = lane & 15, kh = lane >> 4;

    const int ar = tid >> 2, ac = (tid & 3) * 8;            // A: 128x32 bf16
    const int bs = tid >> 7, br = (tid >> 1) & 63,          // B: 2x64x32 f32
              bc = (tid & 1) * 16;
    const float*  Brow  = Bw + (size_t)(n0 + br + (bs ? Nout : 0)) * K + bc;
    const __bf16* Arow0 = Xb + (size_t)(m0 + ar) * K + ac;
    const __bf16* Arow1 = Xb + (size_t)(m0 + ar + 64) * K + ac;

    v8f accG[2][2] = {}, accU[2][2] = {};
    v4f breg[4];

    cp_async_b128(Arow0, &sA[0][ar * LDA_PAD + ac]);
    cp_async_b128(Arow1, &sA[0][(ar + 64) * LDA_PAD + ac]);
#pragma unroll
    for (int q = 0; q < 4; ++q) breg[q] = *(const v4f*)(Brow + 4 * q);

    for (int k0 = 0, it = 0; k0 < K; k0 += BK, ++it) {
        const int buf = it & 1;
        {   // finish staging current buffer (B via regs -> bf16 -> LDS)
            v8bf lo, hi;
            cvt16_bf16(breg, lo, hi);
            __bf16* d = &sB[buf][bs][br * LDB_PAD + bc];
            *(v8bf*)d = lo;
            *(v8bf*)(d + 8) = hi;
        }
        wait_async_lds();
        __syncthreads();
        if (k0 + BK < K) {   // issue next tile into other buffer
            const int nb = buf ^ 1;
            cp_async_b128(Arow0 + k0 + BK, &sA[nb][ar * LDA_PAD + ac]);
            cp_async_b128(Arow1 + k0 + BK, &sA[nb][(ar + 64) * LDA_PAD + ac]);
#pragma unroll
            for (int q = 0; q < 4; ++q)
                breg[q] = *(const v4f*)(Brow + k0 + BK + 4 * q);
            __builtin_prefetch((const void*)(Brow + k0 + 2 * BK), 0, 1);
        }
        v16bf af[2];
#pragma unroll
        for (int i = 0; i < 2; ++i) {
            const __bf16* p = &sA[buf][(wm * 32 + i * 16 + ml) * LDA_PAD + kh * 16];
            ((v8bf*)&af[i])[0] = *(const v8bf*)p;
            ((v8bf*)&af[i])[1] = *(const v8bf*)(p + 8);
        }
#pragma unroll
        for (int j = 0; j < 2; ++j) {
            const int rb = (wn * 32 + j * 16 + ml) * LDB_PAD + kh * 16;
            v16bf bg, bu;
            ((v8bf*)&bg)[0] = *(const v8bf*)&sB[buf][0][rb];
            ((v8bf*)&bg)[1] = *(const v8bf*)&sB[buf][0][rb + 8];
            ((v8bf*)&bu)[0] = *(const v8bf*)&sB[buf][1][rb];
            ((v8bf*)&bu)[1] = *(const v8bf*)&sB[buf][1][rb + 8];
#pragma unroll
            for (int i = 0; i < 2; ++i) {
                accG[i][j] = __builtin_amdgcn_wmma_f32_16x16x32_bf16(
                    false, af[i], false, bg, (short)0, accG[i][j], false, false);
                accU[i][j] = __builtin_amdgcn_wmma_f32_16x16x32_bf16(
                    false, af[i], false, bu, (short)0, accU[i][j], false, false);
            }
        }
    }

    // epilogue: h = scale * silu(g) * u  (C layout: M = r + 8*(lane/16), N = lane%16)
    const int mhi = lane >> 4;
#pragma unroll
    for (int i = 0; i < 2; ++i) {
#pragma unroll
        for (int j = 0; j < 2; ++j) {
            const int col = n0 + wn * 32 + j * 16 + ml;
#pragma unroll
            for (int r = 0; r < 8; ++r) {
                const int row = m0 + wm * 32 + i * 16 + mhi * 8 + r;
                const float g = accG[i][j][r];
                const float u = accU[i][j][r];
                const float s = se ? 1.f : route_w[(size_t)row * 16 + z];
                const float h = s * (g / (1.f + __expf(-g))) * u;
                Out[(size_t)row * Nout + col] = (__bf16)h;
            }
        }
    }
}

// ----------------------- down-proj + combine (one pass) ---------------------
// out[128x128 tile] = Hs @ SW2^T + sum_e Hb_e @ W2_e^T, all in registers.
__global__ __launch_bounds__(256)
void k_down(const __bf16* __restrict__ HbAll, const __bf16* __restrict__ Hs,
            const float* __restrict__ W2, const float* __restrict__ SW2,
            float* __restrict__ out, int T, int H, int I, int Is) {
    const int n0 = blockIdx.x * 128;
    const int m0 = blockIdx.y * 128;

    __shared__ __align__(16) __bf16 sA[2][128 * LDA_PAD];
    __shared__ __align__(16) __bf16 sB[2][128 * LDB_PAD];

    const int tid  = threadIdx.x;
    const int lane = tid & 31, wave = tid >> 5;
    const int wm = wave >> 1, wn = wave & 1;   // wave tile 32 x 64
    const int ml = lane & 15, kh = lane >> 4;
    const int ar = tid >> 2, ac = (tid & 3) * 8;
    const int br = tid >> 1, bc = (tid & 1) * 16;

    v8f acc[2][4] = {};

    for (int seg = 0; seg < 17; ++seg) {
        const bool se = (seg == 16);
        const int K = se ? Is : I;
        const __bf16* A  = se ? Hs  : (HbAll + (size_t)seg * (size_t)T * I);
        const float*  Bw = se ? SW2 : (W2 + (size_t)seg * (size_t)H * I);
        const __bf16* Arow0 = A + (size_t)(m0 + ar) * K + ac;
        const __bf16* Arow1 = A + (size_t)(m0 + ar + 64) * K + ac;
        const float*  Brow  = Bw + (size_t)(n0 + br) * K + bc;

        v4f breg[4];
        cp_async_b128(Arow0, &sA[0][ar * LDA_PAD + ac]);
        cp_async_b128(Arow1, &sA[0][(ar + 64) * LDA_PAD + ac]);
#pragma unroll
        for (int q = 0; q < 4; ++q) breg[q] = *(const v4f*)(Brow + 4 * q);

        for (int k0 = 0, it = 0; k0 < K; k0 += BK, ++it) {
            const int buf = it & 1;
            {
                v8bf lo, hi;
                cvt16_bf16(breg, lo, hi);
                __bf16* d = &sB[buf][br * LDB_PAD + bc];
                *(v8bf*)d = lo;
                *(v8bf*)(d + 8) = hi;
            }
            wait_async_lds();
            __syncthreads();
            if (k0 + BK < K) {
                const int nb = buf ^ 1;
                cp_async_b128(Arow0 + k0 + BK, &sA[nb][ar * LDA_PAD + ac]);
                cp_async_b128(Arow1 + k0 + BK, &sA[nb][(ar + 64) * LDA_PAD + ac]);
#pragma unroll
                for (int q = 0; q < 4; ++q)
                    breg[q] = *(const v4f*)(Brow + k0 + BK + 4 * q);
                __builtin_prefetch((const void*)(Brow + k0 + 2 * BK), 0, 1);
            }
            v16bf af[2];
#pragma unroll
            for (int i = 0; i < 2; ++i) {
                const __bf16* p = &sA[buf][(wm * 32 + i * 16 + ml) * LDA_PAD + kh * 16];
                ((v8bf*)&af[i])[0] = *(const v8bf*)p;
                ((v8bf*)&af[i])[1] = *(const v8bf*)(p + 8);
            }
#pragma unroll
            for (int j = 0; j < 4; ++j) {
                const int rb = (wn * 64 + j * 16 + ml) * LDB_PAD + kh * 16;
                v16bf bf;
                ((v8bf*)&bf)[0] = *(const v8bf*)&sB[buf][rb];
                ((v8bf*)&bf)[1] = *(const v8bf*)&sB[buf][rb + 8];
#pragma unroll
                for (int i = 0; i < 2; ++i)
                    acc[i][j] = __builtin_amdgcn_wmma_f32_16x16x32_bf16(
                        false, af[i], false, bf, (short)0, acc[i][j], false, false);
            }
        }
    }

    const int mhi = lane >> 4;
#pragma unroll
    for (int i = 0; i < 2; ++i) {
#pragma unroll
        for (int j = 0; j < 4; ++j) {
            const int col = n0 + wn * 64 + j * 16 + ml;
#pragma unroll
            for (int r = 0; r < 8; ++r) {
                const int row = m0 + wm * 32 + i * 16 + mhi * 8 + r;
                out[(size_t)row * H + col] = acc[i][j][r];
            }
        }
    }
}

// ------------------------------- launcher ----------------------------------
extern "C" void kernel_launch(void* const* d_in, const int* in_sizes, int n_in,
                              void* d_out, int out_size, void* d_ws, size_t ws_size,
                              hipStream_t stream) {
    const float* X    = (const float*)d_in[0];   // [T,H]
    const float* Gw   = (const float*)d_in[1];   // [E,H]
    const float* bias = (const float*)d_in[2];   // [E]
    const float* W1   = (const float*)d_in[3];   // [E,2I,H]
    const float* W2   = (const float*)d_in[4];   // [E,H,I]
    const float* SW1  = (const float*)d_in[5];   // [2*Is,H]
    const float* SW2  = (const float*)d_in[6];   // [H,Is]
    float* out = (float*)d_out;

    const int T = 1024, H = 2048, I = 1024, E = 16, Is = 2048;

    char* ws = (char*)d_ws;
    __bf16* Xb      = (__bf16*)ws; ws += (size_t)T * H * sizeof(__bf16);          // 4 MB
    float*  route_w = (float*)ws;  ws += (size_t)T * E * sizeof(float);           // 64 KB
    __bf16* HbAll   = (__bf16*)ws; ws += (size_t)E * T * I * sizeof(__bf16);      // 32 MB
    __bf16* Hs      = (__bf16*)ws; ws += (size_t)T * Is * sizeof(__bf16);         // 4 MB

    k_convert<<<(T * H) / 256, 256, 0, stream>>>(X, Xb, T * H);
    k_route<<<T, 32, 0, stream>>>(X, Gw, bias, route_w, T, H);

    dim3 g1(Is / 64, T / 128, E + 1);   // 32 x 8 x 17 (routed z uses x<16)
    k_gateup<<<g1, 256, 0, stream>>>(Xb, W1, SW1, route_w, HbAll, Hs, T, H, I, Is);

    dim3 g2(H / 128, T / 128);          // 16 x 8
    k_down<<<g2, 256, 0, stream>>>(HbAll, Hs, W2, SW2, out, T, H, I, Is);
}
